// NSATransformer_1219770712408
// MI455X (gfx1250) — matrix operand order repaired and checked by reading, not compiled, not run on hardware
//
#include <hip/hip_runtime.h>
#include <math.h>

// ---------------- model constants ----------------
#define SEQ   1024
#define DMOD  512
#define NH    8
#define DHEAD 64
#define FFND  2048
#define NVOC  32000
#define NBLK  32      // SEQ / LBLK
#define LBLK  32
#define NSEL  4
#define WINSZ 64
#define NEGF  (-3.402823466e38f)

typedef __attribute__((ext_vector_type(16))) _Float16 v16h;
typedef __attribute__((ext_vector_type(8)))  float    v8f;

union Frag { v16h h; unsigned int u[8]; };

static __device__ __forceinline__ unsigned int pack2h(float a, float b) {
    union { _Float16 h[2]; unsigned int u; } p;
    p.h[0] = (_Float16)a; p.h[1] = (_Float16)b;
    return p.u;
}

// ====== WMMA GEMM: C[M,N] = A[M,K] @ W[K,N] (+bias, RELU/RESID templated) ======
// Block tile 128x128, 8 waves (2M x 4N), wave tile 64x32 = 8 WMMAs / K-step.
#define BM 128
#define BN 128
#define BK 32
#define LDA (BK + 4)
#define LDB (BK + 4)

template <int RELU, int RESID>
__global__ __launch_bounds__(256) void gemm_wmma_kernel(
    const float* __restrict__ A, const float* __restrict__ W,
    const float* __restrict__ bias, const float* __restrict__ resid,
    float* __restrict__ C, int M, int N, int K)
{
    __shared__ _Float16 As[BM * LDA];   // 9216 B
    __shared__ _Float16 Bs[BN * LDB];   // 9216 B

    const int tid    = threadIdx.x;
    const int lane   = tid & 31;
    const int wave   = tid >> 5;
    const int waveM  = wave >> 2;          // 0..1 -> 64 rows each
    const int waveN  = wave & 3;           // 0..3 -> 32 cols each
    const int blockM = blockIdx.y * BM;
    const int blockN = blockIdx.x * BN;

    v8f acc[4][2] = {};

    // per-lane fragment base offsets (16-bit A/B layouts, CDNA5 ISA)
    const int l15    = lane & 15;
    const int khalf8 = (lane >> 4) * 8;    // A: lanes16-31 hold K+8
    const int kb16   = (lane >> 4) * 16;   // unused: B loads use khalf8 (K+8), same k order as A
    const int mrow0  = (waveM * 64 + l15) * LDA;
    const int nrow0  = (waveN * 32 + l15) * LDB;

    // staging coordinates
    const int arow = tid >> 1;             // 0..127
    const int acol = (tid & 1) * 16;       // 0/16
    const int bkr  = (tid >> 4) * 2;       // 0,2,..,30 (k pair)
    const int bcol = (tid & 15) * 8;       // 0..120

    for (int k0 = 0; k0 < K; k0 += BK) {
        // ---- stage A tile 128x32 fp32->f16 (4x b128 loads, 8x b32 LDS stores) ----
        {
            const float4* src = (const float4*)(A + (size_t)(blockM + arow) * K + k0 + acol);
            const float4 f0 = src[0], f1 = src[1], f2 = src[2], f3 = src[3];
            unsigned int* dst = (unsigned int*)&As[arow * LDA + acol];
            dst[0] = pack2h(f0.x, f0.y);
            dst[1] = pack2h(f0.z, f0.w);
            dst[2] = pack2h(f1.x, f1.y);
            dst[3] = pack2h(f1.z, f1.w);
            dst[4] = pack2h(f2.x, f2.y);
            dst[5] = pack2h(f2.z, f2.w);
            dst[6] = pack2h(f3.x, f3.y);
            dst[7] = pack2h(f3.z, f3.w);
        }
        // ---- stage B tile 32x128 transposed -> Bs[n][k] (4x b128 loads, 8x b32) ----
        {
            const float4* s0 = (const float4*)(W + (size_t)(k0 + bkr)     * N + blockN + bcol);
            const float4* s1 = (const float4*)(W + (size_t)(k0 + bkr + 1) * N + blockN + bcol);
            const float4 r0a = s0[0], r0b = s0[1];
            const float4 r1a = s1[0], r1b = s1[1];
            const float ra[8] = {r0a.x, r0a.y, r0a.z, r0a.w, r0b.x, r0b.y, r0b.z, r0b.w};
            const float rb[8] = {r1a.x, r1a.y, r1a.z, r1a.w, r1b.x, r1b.y, r1b.z, r1b.w};
            #pragma unroll
            for (int j = 0; j < 8; ++j)
                *(unsigned int*)&Bs[(bcol + j) * LDB + bkr] = pack2h(ra[j], rb[j]);
        }
        // prefetch next K tile (global_prefetch_b8) while this one is consumed
        if (k0 + BK < K) {
            __builtin_prefetch(A + (size_t)(blockM + arow) * K + k0 + BK + acol, 0, 1);
            __builtin_prefetch(W + (size_t)(k0 + BK + bkr) * N + blockN + bcol, 0, 1);
        }
        __syncthreads();

        Frag af[4], bf[2];
        #pragma unroll
        for (int sm = 0; sm < 4; ++sm) {
            const unsigned int* p0 = (const unsigned int*)&As[mrow0 + sm * 16 * LDA + khalf8];
            const unsigned int* p1 = (const unsigned int*)&As[mrow0 + sm * 16 * LDA + 16 + khalf8];
            #pragma unroll
            for (int i = 0; i < 4; ++i) { af[sm].u[i] = p0[i]; af[sm].u[4 + i] = p1[i]; }
        }
        #pragma unroll
        for (int sn = 0; sn < 2; ++sn) {
            const unsigned int* pb = (const unsigned int*)&Bs[nrow0 + sn * 16 * LDB + khalf8];
            const unsigned int* pb1 = (const unsigned int*)&Bs[nrow0 + sn * 16 * LDB + 16 + khalf8];
            #pragma unroll
            for (int i = 0; i < 4; ++i) { bf[sn].u[i] = pb[i]; bf[sn].u[4 + i] = pb1[i]; }
        }
        #pragma unroll
        for (int sm = 0; sm < 4; ++sm)
            #pragma unroll
            for (int sn = 0; sn < 2; ++sn)
                acc[sm][sn] = __builtin_amdgcn_wmma_f32_16x16x32_f16(
                    false, af[sm].h, false, bf[sn].h,
                    (short)0, acc[sm][sn], false, false);
        __syncthreads();
    }

    // epilogue: C/D layout -> lane l holds N=(l&15), rows r + (l>>4)*8. Branch-free.
    const int rbase = (lane >> 4) * 8;
    #pragma unroll
    for (int sm = 0; sm < 4; ++sm) {
        #pragma unroll
        for (int sn = 0; sn < 2; ++sn) {
            const int col  = blockN + waveN * 32 + sn * 16 + l15;
            const int row0 = blockM + waveM * 64 + sm * 16 + rbase;
            const float bcolv = bias[col];
            float rres[8];
            if (RESID) {
                #pragma unroll
                for (int r = 0; r < 8; ++r)
                    rres[r] = resid[(size_t)(row0 + r) * N + col];
            }
            #pragma unroll
            for (int r = 0; r < 8; ++r) {
                float v = acc[sm][sn][r] + bcolv;
                if (RELU)  v = fmaxf(v, 0.0f);
                if (RESID) v += rres[r];
                C[(size_t)(row0 + r) * N + col] = v;
            }
        }
    }
}

// ================= embedding ===========================================
__global__ __launch_bounds__(256) void embed_kernel(
    const int* __restrict__ tokens, const float* __restrict__ tok_emb,
    const float* __restrict__ pos_emb, float* __restrict__ x)
{
    const int s = blockIdx.x, t = threadIdx.x;
    const int tok = tokens[s];
    #pragma unroll
    for (int j = 0; j < 2; ++j) {
        const int d = t + j * 256;
        x[s * DMOD + d] = tok_emb[(size_t)tok * DMOD + d] + pos_emb[s * DMOD + d];
    }
}

// ================= layernorm (block per row) ===========================
__global__ __launch_bounds__(256) void ln_kernel(
    const float* __restrict__ x, const float* __restrict__ g,
    const float* __restrict__ b, float* __restrict__ y)
{
    __shared__ float red[256];
    const int s = blockIdx.x, t = threadIdx.x;
    const float v0 = x[s * DMOD + t], v1 = x[s * DMOD + t + 256];
    red[t] = v0 + v1; __syncthreads();
    for (int off = 128; off > 0; off >>= 1) { if (t < off) red[t] += red[t + off]; __syncthreads(); }
    const float mean = red[0] * (1.0f / DMOD);
    __syncthreads();
    const float d0 = v0 - mean, d1 = v1 - mean;
    red[t] = d0 * d0 + d1 * d1; __syncthreads();
    for (int off = 128; off > 0; off >>= 1) { if (t < off) red[t] += red[t + off]; __syncthreads(); }
    const float inv = rsqrtf(red[0] * (1.0f / DMOD) + 1e-5f);
    y[s * DMOD + t]       = d0 * inv * g[t]       + b[t];
    y[s * DMOD + t + 256] = d1 * inv * g[t + 256] + b[t + 256];
}

// ================= compress: block means of k,v ========================
__global__ __launch_bounds__(64) void compress_kernel(
    const float* __restrict__ k, const float* __restrict__ v,
    float* __restrict__ kcmp, float* __restrict__ vcmp)
{
    const int blk = blockIdx.x;          // h*NBLK + n  (256 blocks)
    const int h = blk >> 5, n = blk & 31;
    const int dh = threadIdx.x;
    float sk = 0.f, sv = 0.f;
    for (int j = 0; j < LBLK; ++j) {
        const int row = n * LBLK + j;
        sk += k[row * DMOD + h * DHEAD + dh];
        sv += v[row * DMOD + h * DHEAD + dh];
    }
    kcmp[blk * DHEAD + dh] = sk * (1.0f / LBLK);
    vcmp[blk * DHEAD + dh] = sv * (1.0f / LBLK);
}

// ================= compressed attention + top-4 (wave per (s,h)) =======
__global__ __launch_bounds__(256) void cmp_attn_kernel(
    const float* __restrict__ q, const float* __restrict__ kcmp,
    const float* __restrict__ vcmp, float* __restrict__ out, int* __restrict__ idx)
{
    const int lane = threadIdx.x;                  // = block index n (0..31)
    const int pair = blockIdx.x * 8 + threadIdx.y; // s*NH + h
    const int s = pair >> 3, h = pair & 7;

    const float* qr = q + s * DMOD + h * DHEAD;
    const float* kr = kcmp + (h * NBLK + lane) * DHEAD;
    float dot = 0.f;
    for (int d = 0; d < DHEAD; ++d) dot += qr[d] * kr[d];
    const bool valid = (lane * LBLK + (LBLK - 1)) <= s;
    const float sc = valid ? dot * 0.125f : NEGF;

    // softmax over 32 lanes
    float m = sc;
    for (int o = 16; o > 0; o >>= 1) m = fmaxf(m, __shfl_xor(m, o, 32));
    float e = valid ? __expf(sc - m) : 0.0f;
    float sum = e;
    for (int o = 16; o > 0; o >>= 1) sum += __shfl_xor(sum, o, 32);
    const float p = e / fmaxf(sum, 1e-9f);

    // top-4 (tie -> smallest index, matching jax.lax.top_k; removed -> -inf)
    float val = sc;
    #pragma unroll
    for (int t = 0; t < NSEL; ++t) {
        float bv = val; int bi = lane;
        for (int o = 16; o > 0; o >>= 1) {
            float ov = __shfl_xor(bv, o, 32);
            int   oi = __shfl_xor(bi, o, 32);
            if (ov > bv || (ov == bv && oi < bi)) { bv = ov; bi = oi; }
        }
        if (lane == 0) idx[pair * NSEL + t] = bi;
        if (lane == bi) val = -INFINITY;
    }

    // weighted sum over v_cmp: lane owns dims (lane, lane+32)
    float a0 = 0.f, a1 = 0.f;
    for (int n = 0; n < NBLK; ++n) {
        const float pn = __shfl(p, n, 32);
        const float* vr = vcmp + (h * NBLK + n) * DHEAD;
        a0 += pn * vr[lane];
        a1 += pn * vr[lane + 32];
    }
    out[s * DMOD + h * DHEAD + lane]      = a0;
    out[s * DMOD + h * DHEAD + lane + 32] = a1;
}

// ================= selected-block attention (wave per (s,h)) ===========
__global__ __launch_bounds__(256) void sel_attn_kernel(
    const float* __restrict__ q, const float* __restrict__ k,
    const float* __restrict__ v, const int* __restrict__ idx,
    float* __restrict__ out)
{
    __shared__ float pbuf[8][NSEL * LBLK];
    const int lane = threadIdx.x, ty = threadIdx.y;
    const int pair = blockIdx.x * 8 + ty;
    const int s = pair >> 3, h = pair & 7;

    int bidx[NSEL];
    #pragma unroll
    for (int t = 0; t < NSEL; ++t) bidx[t] = idx[pair * NSEL + t];

    const float* qr = q + s * DMOD + h * DHEAD;
    float sc[NSEL]; bool vmask[NSEL];
    #pragma unroll
    for (int t = 0; t < NSEL; ++t) {
        const int row = bidx[t] * LBLK + lane;
        vmask[t] = row <= s;
        const float* kr = k + row * DMOD + h * DHEAD;
        float dot = 0.f;
        for (int d = 0; d < DHEAD; ++d) dot += qr[d] * kr[d];
        sc[t] = vmask[t] ? dot * 0.125f : NEGF;
    }
    float m = fmaxf(fmaxf(sc[0], sc[1]), fmaxf(sc[2], sc[3]));
    for (int o = 16; o > 0; o >>= 1) m = fmaxf(m, __shfl_xor(m, o, 32));
    float e[NSEL], sum = 0.f;
    #pragma unroll
    for (int t = 0; t < NSEL; ++t) { e[t] = vmask[t] ? __expf(sc[t] - m) : 0.0f; sum += e[t]; }
    for (int o = 16; o > 0; o >>= 1) sum += __shfl_xor(sum, o, 32);
    const float inv = 1.0f / fmaxf(sum, 1e-9f);
    #pragma unroll
    for (int t = 0; t < NSEL; ++t) pbuf[ty][t * 32 + lane] = e[t] * inv;
    __syncthreads();

    float a0 = 0.f, a1 = 0.f;
    for (int kk = 0; kk < NSEL * LBLK; ++kk) {
        const int row = bidx[kk >> 5] * LBLK + (kk & 31);
        const float p = pbuf[ty][kk];
        const float* vr = v + row * DMOD + h * DHEAD;
        a0 += p * vr[lane];
        a1 += p * vr[lane + 32];
    }
    out[s * DMOD + h * DHEAD + lane]      = a0;
    out[s * DMOD + h * DHEAD + lane + 32] = a1;
}

// ================= sliding-window attention (wave per (s,h)) ===========
__global__ __launch_bounds__(256) void win_attn_kernel(
    const float* __restrict__ q, const float* __restrict__ k,
    const float* __restrict__ v, float* __restrict__ out)
{
    __shared__ float pbuf[8][WINSZ];
    const int lane = threadIdx.x, ty = threadIdx.y;
    const int pair = blockIdx.x * 8 + ty;
    const int s = pair >> 3, h = pair & 7;

    const float* qr = q + s * DMOD + h * DHEAD;
    float sc[2]; bool vmask[2];
    #pragma unroll
    for (int t = 0; t < 2; ++t) {
        const int kk = t * 32 + lane;
        const int wpos = s - (WINSZ - 1) + kk;
        vmask[t] = wpos >= 0;
        const int row = min(max(wpos, 0), SEQ - 1);
        const float* kr = k + row * DMOD + h * DHEAD;
        float dot = 0.f;
        for (int d = 0; d < DHEAD; ++d) dot += qr[d] * kr[d];
        sc[t] = vmask[t] ? dot * 0.125f : NEGF;
    }
    float m = fmaxf(sc[0], sc[1]);
    for (int o = 16; o > 0; o >>= 1) m = fmaxf(m, __shfl_xor(m, o, 32));
    float e[2], sum = 0.f;
    #pragma unroll
    for (int t = 0; t < 2; ++t) { e[t] = vmask[t] ? __expf(sc[t] - m) : 0.0f; sum += e[t]; }
    for (int o = 16; o > 0; o >>= 1) sum += __shfl_xor(sum, o, 32);
    const float inv = 1.0f / fmaxf(sum, 1e-9f);
    #pragma unroll
    for (int t = 0; t < 2; ++t) pbuf[ty][t * 32 + lane] = e[t] * inv;
    __syncthreads();

    float a0 = 0.f, a1 = 0.f;
    for (int kk = 0; kk < WINSZ; ++kk) {
        const int wpos = s - (WINSZ - 1) + kk;
        const int row = min(max(wpos, 0), SEQ - 1);
        const float p = pbuf[ty][kk];
        const float* vr = v + row * DMOD + h * DHEAD;
        a0 += p * vr[lane];
        a1 += p * vr[lane + 32];
    }
    out[s * DMOD + h * DHEAD + lane]      = a0;
    out[s * DMOD + h * DHEAD + lane + 32] = a1;
}

// ================= gates: sigmoid(xn @ wg + bg), wave per gate =========
__global__ __launch_bounds__(256) void gate_kernel(
    const float* __restrict__ xn, const float* __restrict__ wg,
    const float* __restrict__ bg, float* __restrict__ gates)
{
    const int s = blockIdx.x;
    const int wave = threadIdx.x >> 5, lane = threadIdx.x & 31;
    for (int j = wave; j < 3 * NH; j += 8) {
        float dot = 0.f;
        for (int t = lane; t < DMOD; t += 32)
            dot += xn[s * DMOD + t] * wg[t * (3 * NH) + j];
        for (int o = 16; o > 0; o >>= 1) dot += __shfl_xor(dot, o, 32);
        if (lane == 0)
            gates[s * (3 * NH) + j] = 1.0f / (1.0f + __expf(-(dot + bg[j])));
    }
}

// ================= gated combine =======================================
__global__ __launch_bounds__(256) void combine_kernel(
    const float* __restrict__ gates, const float* __restrict__ ocmp,
    const float* __restrict__ osel, const float* __restrict__ owin,
    float* __restrict__ comb)
{
    const int s = blockIdx.x, t = threadIdx.x;
    #pragma unroll
    for (int j = 0; j < 2; ++j) {
        const int d = t + j * 256;
        const int h = d >> 6;
        const float g0 = gates[s * 24 + h * 3 + 0];
        const float g1 = gates[s * 24 + h * 3 + 1];
        const float g2 = gates[s * 24 + h * 3 + 2];
        const size_t o = (size_t)s * DMOD + d;
        comb[o] = g0 * ocmp[o] + g1 * osel[o] + g2 * owin[o];
    }
}

// ================= host orchestration ==================================
extern "C" void kernel_launch(void* const* d_in, const int* in_sizes, int n_in,
                              void* d_out, int out_size, void* d_ws, size_t ws_size,
                              hipStream_t stream) {
    (void)in_sizes; (void)n_in; (void)out_size; (void)ws_size;
    const int*   tokens  = (const int*)d_in[0];
    const float* tok_emb = (const float*)d_in[1];
    const float* pos_emb = (const float*)d_in[2];
    const float* ln1_g = (const float*)d_in[3];
    const float* ln1_b = (const float*)d_in[4];
    const float* ln2_g = (const float*)d_in[5];
    const float* ln2_b = (const float*)d_in[6];
    const float* wq = (const float*)d_in[7];
    const float* bq = (const float*)d_in[8];
    const float* wk = (const float*)d_in[9];
    const float* bk = (const float*)d_in[10];
    const float* wv = (const float*)d_in[11];
    const float* bv = (const float*)d_in[12];
    const float* wo = (const float*)d_in[13];
    const float* bo = (const float*)d_in[14];
    const float* wg = (const float*)d_in[15];
    const float* bg = (const float*)d_in[16];
    const float* w1 = (const float*)d_in[17];
    const float* b1 = (const float*)d_in[18];
    const float* w2 = (const float*)d_in[19];
    const float* b2 = (const float*)d_in[20];
    const float* norm_g = (const float*)d_in[21];
    const float* norm_b = (const float*)d_in[22];
    const float* fc_w   = (const float*)d_in[23];
    const float* fc_b   = (const float*)d_in[24];

    float* ws   = (float*)d_ws;
    float* x    = ws;
    float* xn   = x    + SEQ * DMOD;
    float* qb   = xn   + SEQ * DMOD;
    float* kb   = qb   + SEQ * DMOD;
    float* vb   = kb   + SEQ * DMOD;
    float* ocmp = vb   + SEQ * DMOD;
    float* osel = ocmp + SEQ * DMOD;
    float* owin = osel + SEQ * DMOD;
    float* comb = owin + SEQ * DMOD;
    float* hbuf = comb + SEQ * DMOD;          // SEQ*FFND
    float* kcmp = hbuf + SEQ * FFND;
    float* vcmp = kcmp + NH * NBLK * DHEAD;
    float* gts  = vcmp + NH * NBLK * DHEAD;   // SEQ*24
    int*   idxb = (int*)(gts + SEQ * 24);     // SEQ*NH*NSEL

    const dim3 blk256(256);
    const dim3 blkAttn(32, 8);
    const dim3 gD(DMOD / BN, SEQ / BM);       // (4,8)
    const dim3 gF(FFND / BN, SEQ / BM);       // (16,8)
    const dim3 gV(NVOC / BN, SEQ / BM);       // (250,8)

    embed_kernel<<<SEQ, blk256, 0, stream>>>(tokens, tok_emb, pos_emb, x);

    for (int l = 0; l < 2; ++l) {
        const size_t oD = (size_t)l * DMOD * DMOD;
        const size_t oF = (size_t)l * DMOD * FFND;
        ln_kernel<<<SEQ, blk256, 0, stream>>>(x, ln1_g + l * DMOD, ln1_b + l * DMOD, xn);
        gemm_wmma_kernel<0,0><<<gD, blk256, 0, stream>>>(xn, wq + oD, bq + l * DMOD, nullptr, qb, SEQ, DMOD, DMOD);
        gemm_wmma_kernel<0,0><<<gD, blk256, 0, stream>>>(xn, wk + oD, bk + l * DMOD, nullptr, kb, SEQ, DMOD, DMOD);
        gemm_wmma_kernel<0,0><<<gD, blk256, 0, stream>>>(xn, wv + oD, bv + l * DMOD, nullptr, vb, SEQ, DMOD, DMOD);
        compress_kernel<<<NH * NBLK, 64, 0, stream>>>(kb, vb, kcmp, vcmp);
        cmp_attn_kernel<<<SEQ, blkAttn, 0, stream>>>(qb, kcmp, vcmp, ocmp, idxb);
        sel_attn_kernel<<<SEQ, blkAttn, 0, stream>>>(qb, kb, vb, idxb, osel);
        win_attn_kernel<<<SEQ, blkAttn, 0, stream>>>(qb, kb, vb, owin);
        gate_kernel<<<SEQ, blk256, 0, stream>>>(xn, wg + (size_t)l * DMOD * 24, bg + l * 24, gts);
        combine_kernel<<<SEQ, blk256, 0, stream>>>(gts, ocmp, osel, owin, comb);
        gemm_wmma_kernel<0,1><<<gD, blk256, 0, stream>>>(comb, wo + oD, bo + l * DMOD, x, x, SEQ, DMOD, DMOD);
        ln_kernel<<<SEQ, blk256, 0, stream>>>(x, ln2_g + l * DMOD, ln2_b + l * DMOD, xn);
        gemm_wmma_kernel<1,0><<<gF, blk256, 0, stream>>>(xn, w1 + oF, b1 + l * FFND, nullptr, hbuf, SEQ, FFND, DMOD);
        gemm_wmma_kernel<0,1><<<gD, blk256, 0, stream>>>(hbuf, w2 + oF, b2 + l * DMOD, x, x, SEQ, DMOD, FFND);
    }

    ln_kernel<<<SEQ, blk256, 0, stream>>>(x, norm_g, norm_b, xn);
    gemm_wmma_kernel<0,0><<<gV, blk256, 0, stream>>>(xn, fc_w, fc_b, nullptr, (float*)d_out, SEQ, NVOC, DMOD);
}
